// SynapticStorage_70068096467168
// MI455X (gfx1250) — matrix operand order repaired and checked by reading, not compiled, not run on hardware
//
#include <hip/hip_runtime.h>
#include <hip/hip_bf16.h>

// ---------------------------------------------------------------------------
// SynapticStorage retrieval for MI455X (gfx1250, wave32, WMMA)
//   1) row_inv_norm      : 1/max(||p_c||,eps) for all (padded) pattern rows
//   2) q_norm_bf16       : normalized queries -> bf16 row-major [B][D]
//   3) p_transpose_bf16  : normalized patterns -> bf16 row-major [Cpad][D]
//   4) gemm_topk         : bf16 WMMA 16x16x32 GEMM, dual accumulator chains,
//                          1-deep B prefetch pipeline, fused per-row top-8
//                          (64 query slabs x 64 pattern chunks of 800)
//   5) merge_out         : merge 64x8 partials -> top-8, weighted softmax,
//                          weighted combine of original fp32 patterns
// Workspace: ~58 MB (see layout in kernel_launch).
// ---------------------------------------------------------------------------

#define DDIM   512
#define BQ     1024
#define CREAL  50000
#define CPAD   51200          // 64 chunks * 800, multiple of 64
#define CHUNK  800            // multiple of 32
#define NCHUNK 64
#define NPAIR  (CHUNK / 32)   // 25 tile-pairs (50 16-wide tiles)
#define KSTEPS (DDIM / 32)    // 16
#define NSLAB  (BQ / 16)      // 64
#define NEGINF (-3.4e38f)

typedef __attribute__((ext_vector_type(16))) __bf16       v16bf;
typedef __attribute__((ext_vector_type(8)))  float        v8f;
typedef __attribute__((ext_vector_type(4)))  unsigned int u32x4;

union Frag { v16bf v; u32x4 u[2]; };

__device__ __forceinline__ unsigned short f2bf(float f) {
    unsigned int u = __float_as_uint(f);
    unsigned int r = u + 0x7FFFu + ((u >> 16) & 1u);   // round-to-nearest-even
    return (unsigned short)(r >> 16);
}

// --- 1) per-row inverse L2 norm (rows of a [*][512] fp32 matrix) -----------
__global__ __launch_bounds__(256) void row_inv_norm(
    const float* __restrict__ in, float* __restrict__ inv, int valid)
{
    int row = blockIdx.x, tid = threadIdx.x;
    __shared__ float red[256];
    float s = 0.f;
    if (row < valid) {
        float x0 = in[(size_t)row * DDIM + tid];
        float x1 = in[(size_t)row * DDIM + tid + 256];
        s = x0 * x0 + x1 * x1;
    }
    red[tid] = s; __syncthreads();
    for (int st = 128; st > 0; st >>= 1) {
        if (tid < st) red[tid] += red[tid + st];
        __syncthreads();
    }
    if (tid == 0)
        inv[row] = (row < valid) ? (1.0f / fmaxf(sqrtf(red[0]), 1e-8f)) : 0.f;
}

// --- 2) normalize queries, write bf16 row-major ----------------------------
__global__ __launch_bounds__(256) void q_norm_bf16(
    const float* __restrict__ q, unsigned short* __restrict__ qn)
{
    int row = blockIdx.x, tid = threadIdx.x;
    __shared__ float red[256];
    __shared__ float sinv;
    float x0 = q[(size_t)row * DDIM + tid];
    float x1 = q[(size_t)row * DDIM + tid + 256];
    red[tid] = x0 * x0 + x1 * x1; __syncthreads();
    for (int st = 128; st > 0; st >>= 1) {
        if (tid < st) red[tid] += red[tid + st];
        __syncthreads();
    }
    if (tid == 0) sinv = 1.0f / fmaxf(sqrtf(red[0]), 1e-8f);
    __syncthreads();
    qn[(size_t)row * DDIM + tid]       = f2bf(x0 * sinv);
    qn[(size_t)row * DDIM + tid + 256] = f2bf(x1 * sinv);
}

// --- 3) normalize patterns via LDS: [Cpad][D] fp32 -> [Cpad][D] bf16 ---
__global__ __launch_bounds__(256) void p_transpose_bf16(
    const float* __restrict__ p, const float* __restrict__ pinv,
    unsigned short* __restrict__ pnT)
{
    const int c0 = (blockIdx.x % (CPAD / 64)) * 64;
    const int k0 = (blockIdx.x / (CPAD / 64)) * 64;
    __shared__ unsigned short lds[64 * 65];
    const int tid = threadIdx.x;
    for (int e = tid; e < 64 * 64; e += 256) {
        int i = e >> 6, k = e & 63;
        int c = c0 + i;
        float v = (c < CREAL) ? p[(size_t)c * DDIM + k0 + k] * pinv[c] : 0.f;
        lds[i * 65 + k] = f2bf(v);
    }
    __syncthreads();
    for (int e = tid; e < 64 * 64; e += 256) {
        int j = e >> 6, i = e & 63;
        pnT[(size_t)(c0 + i) * DDIM + k0 + j] = lds[i * 65 + j];
    }
}

// --- 4) one wave: 16 query rows x 800-pattern chunk; WMMA + fused top-8 ----
__global__ __launch_bounds__(32) void gemm_topk(
    const unsigned short* __restrict__ qn,    // [BQ][D]   bf16, normalized
    const unsigned short* __restrict__ pnT,   // [CPAD][D] bf16, normalized
    float* __restrict__ pvals, int* __restrict__ pidx)
{
    const int slab  = blockIdx.x / NCHUNK;
    const int chunk = blockIdx.x % NCHUNK;
    const int l     = threadIdx.x;

    __shared__ float tile[16 * 32];
    __shared__ float topv[16 * 8];
    __shared__ int   topi[16 * 8];
    for (int i = l; i < 128; i += 32) { topv[i] = NEGINF; topi[i] = 0; }

    // Preload all 16 A fragments for this 16-row query slab into VGPRs.
    // A layout (16-bit 16x32): lane&15 = row; halves = K (l>>4 selects 8-K group)
    Frag a[KSTEPS];
    const unsigned short* qrow = qn + (size_t)(slab * 16 + (l & 15)) * DDIM;
    const int ksel = (l >> 4) << 3;
#pragma unroll
    for (int t = 0; t < KSTEPS; ++t) {
        a[t].u[0] = *(const u32x4*)(qrow + t * 32 + ksel);
        a[t].u[1] = *(const u32x4*)(qrow + t * 32 + 16 + ksel);
    }
    __syncthreads();

    for (int pr = 0; pr < NPAIR; ++pr) {
        const int pbase = chunk * CHUNK + pr * 32;   // two adjacent 16-tiles
        // B layout (16-bit 32x16): lane&15 = N; halves = K (l>>4 selects 8-K group), same as A.
        // pnT is [CPAD][D]; each lane reads its pattern row along K.
        const unsigned short* brow = pnT + (size_t)(pbase + (l & 15)) * DDIM + ksel;

        v8f acc0 = {0.f, 0.f, 0.f, 0.f, 0.f, 0.f, 0.f, 0.f};
        v8f acc1 = acc0;

        Frag b0c, b1c, b0n, b1n;
        b0c.u[0] = *(const u32x4*)(brow);
        b0c.u[1] = *(const u32x4*)(brow + 16);
        b1c.u[0] = *(const u32x4*)(brow + 16 * DDIM);
        b1c.u[1] = *(const u32x4*)(brow + 16 * DDIM + 16);
#pragma unroll
        for (int t = 0; t < KSTEPS; ++t) {
            if (t + 1 < KSTEPS) {   // prefetch next k-step while computing this one
                const unsigned short* bn = brow + (size_t)(t + 1) * 32;
                b0n.u[0] = *(const u32x4*)(bn);
                b0n.u[1] = *(const u32x4*)(bn + 16);
                b1n.u[0] = *(const u32x4*)(bn + 16 * DDIM);
                b1n.u[1] = *(const u32x4*)(bn + 16 * DDIM + 16);
            }
            acc0 = __builtin_amdgcn_wmma_f32_16x16x32_bf16(
                false, a[t].v, false, b0c.v, (short)0, acc0, false, false);
            acc1 = __builtin_amdgcn_wmma_f32_16x16x32_bf16(
                false, a[t].v, false, b1c.v, (short)0, acc1, false, false);
            b0c = b0n; b1c = b1n;
        }

        // C/D layout: VGPR v, lanes 0-15 -> M=v; lanes 16-31 -> M=v+8; N=l&15
        const int col = l & 15, rsel = (l >> 4) * 8;
#pragma unroll
        for (int v = 0; v < 8; ++v) {
            tile[(rsel + v) * 32 + col]      = acc0[v];
            tile[(rsel + v) * 32 + 16 + col] = acc1[v];
        }
        __syncthreads();

        if (l < 16) {  // lane r maintains sorted-descending top-8 of query row r
            float tmin = topv[l * 8 + 7];
            for (int n = 0; n < 32; ++n) {
                float val = tile[l * 32 + n];
                int   g   = pbase + n;
                if (g < CREAL && val > tmin) {
                    int j = 7;
                    while (j > 0 && val > topv[l * 8 + j - 1]) {
                        topv[l * 8 + j] = topv[l * 8 + j - 1];
                        topi[l * 8 + j] = topi[l * 8 + j - 1];
                        --j;
                    }
                    topv[l * 8 + j] = val;
                    topi[l * 8 + j] = g;
                    tmin = topv[l * 8 + 7];
                }
            }
        }
        __syncthreads();
    }

    if (l < 16) {
        const int b = slab * 16 + l;
        const size_t base = ((size_t)b * NCHUNK + chunk) * 8;
        for (int j = 0; j < 8; ++j) {
            pvals[base + j] = topv[l * 8 + j];
            pidx [base + j] = topi[l * 8 + j];
        }
    }
}

// --- 5) merge partials -> top-8, weighted softmax, combine patterns --------
__global__ __launch_bounds__(256) void merge_out(
    const float* __restrict__ pvals, const int* __restrict__ pidx,
    const float* __restrict__ patterns, const float* __restrict__ cplx,
    float* __restrict__ out)
{
    const int b = blockIdx.x, tid = threadIdx.x;
    __shared__ float cv[512];   __shared__ int ci[512];
    __shared__ float redv[256]; __shared__ int redp[256];
    __shared__ float selv[8];   __shared__ int seli[8];

    for (int i = tid; i < 512; i += 256) {
        cv[i] = pvals[(size_t)b * 512 + i];
        ci[i] = pidx [(size_t)b * 512 + i];
    }
    __syncthreads();

    for (int k = 0; k < 8; ++k) {
        float best = NEGINF; int bp = 0;
        for (int i = tid; i < 512; i += 256)
            if (cv[i] > best) { best = cv[i]; bp = i; }
        redv[tid] = best; redp[tid] = bp;
        __syncthreads();
        for (int s = 128; s > 0; s >>= 1) {
            if (tid < s && redv[tid + s] > redv[tid]) {
                redv[tid] = redv[tid + s]; redp[tid] = redp[tid + s];
            }
            __syncthreads();
        }
        if (tid == 0) { int p = redp[0]; selv[k] = redv[0]; seli[k] = ci[p]; cv[p] = NEGINF; }
        __syncthreads();
    }

    float w[8]; float m = NEGINF;
#pragma unroll
    for (int k = 0; k < 8; ++k) {
        float v = selv[k] * (1.f + cplx[seli[k]]);
        w[k] = v; m = fmaxf(m, v);
    }
    float s = 0.f;
#pragma unroll
    for (int k = 0; k < 8; ++k) { w[k] = __expf(w[k] - m); s += w[k]; }
    const float invs = 1.f / s;

    for (int d = tid; d < DDIM; d += 256) {
        float acc = 0.f;
#pragma unroll
        for (int k = 0; k < 8; ++k)
            acc += w[k] * patterns[(size_t)seli[k] * DDIM + d];
        out[(size_t)b * DDIM + d] = acc * invs;
    }
}

// ---------------------------------------------------------------------------
extern "C" void kernel_launch(void* const* d_in, const int* in_sizes, int n_in,
                              void* d_out, int out_size, void* d_ws, size_t ws_size,
                              hipStream_t stream)
{
    const float* q    = (const float*)d_in[0];   // [1024][512]
    const float* p    = (const float*)d_in[1];   // [50000][512]
    const float* cplx = (const float*)d_in[2];   // [50000]
    // d_in[3] = top_k (==8, hardcoded)
    float* out = (float*)d_out;

    // Workspace layout (bytes), total ~57.9 MB:
    char* ws = (char*)d_ws;
    size_t off = 0;
    unsigned short* pnT = (unsigned short*)(ws + off); off += (size_t)CPAD * DDIM * 2;  // 52,428,800
    unsigned short* qn  = (unsigned short*)(ws + off); off += (size_t)BQ   * DDIM * 2;  //  1,048,576
    float* pinv  = (float*)(ws + off); off += (size_t)CPAD * 4;                          //    204,800
    float* pvals = (float*)(ws + off); off += (size_t)BQ * NCHUNK * 8 * 4;               //  2,097,152
    int*   pidx  = (int*)  (ws + off); off += (size_t)BQ * NCHUNK * 8 * 4;               //  2,097,152
    (void)ws_size; (void)in_sizes; (void)n_in; (void)out_size;

    row_inv_norm    <<<CPAD, 256, 0, stream>>>(p, pinv, CREAL);
    q_norm_bf16     <<<BQ,   256, 0, stream>>>(q, qn);
    p_transpose_bf16<<<(CPAD / 64) * (DDIM / 64), 256, 0, stream>>>(p, pinv, pnT);
    gemm_topk       <<<NSLAB * NCHUNK, 32, 0, stream>>>(qn, pnT, pvals, pidx);
    merge_out       <<<BQ,   256, 0, stream>>>(pvals, pidx, p, cplx, out);
}
